// RegularizedParallelRNNet_4664334483573
// MI455X (gfx1250) — matrix operand order repaired and checked by reading, not compiled, not run on hardware
//
#include <hip/hip_runtime.h>
#include <cstdint>

typedef __attribute__((ext_vector_type(16))) _Float16 v16h;
typedef __attribute__((ext_vector_type(8)))  _Float16 v8h;
typedef __attribute__((ext_vector_type(8)))  float    v8f;
typedef __attribute__((ext_vector_type(4)))  float    v4f;

#define DIMS 32
#define HID  16
#define TT   512
#define ROWH 24   // halfword stride per 16-wide h row: 48B -> 16B aligned, bank-conflict free

__device__ __forceinline__ float fast_tanh(float x) {
#if __has_builtin(__builtin_amdgcn_tanhf)
  return __builtin_amdgcn_tanhf(x);           // gfx1250 native V_TANH_F32 (1 trans op)
#else
  float ax = __builtin_fabsf(x);
  float e  = __builtin_amdgcn_exp2f(ax * -2.88539008f);   // exp(-2|x|)
  float r  = (1.0f - e) * __builtin_amdgcn_rcpf(1.0f + e);
  return __builtin_copysignf(r, x);
#endif
}

union V16 { v16h v; v8h h[2]; };

// ---------------------------------------------------------------------------
// Kernel 2: the WMMA recurrence. One wave32 owns one (dim d, 16-batch tile).
// z^T formulation: D[M=k_out, N=b] = A(weights) x B(h^T) + C(input/bias).
// ---------------------------------------------------------------------------
__global__ __launch_bounds__(128) void rnn_wmma_kernel(
    const float* __restrict__ x, const float* __restrict__ stats,
    const float* __restrict__ Wih0, const float* __restrict__ Whh0,
    const float* __restrict__ bih0, const float* __restrict__ bhh0,
    const float* __restrict__ Wih1, const float* __restrict__ Whh1,
    const float* __restrict__ bih1, const float* __restrict__ bhh1,
    float* __restrict__ feat)
{
  __shared__ __align__(16) _Float16 hbuf[4][2][16 * ROWH];  // per-wave h0/h1 rows [b][k], f16
  const int wv   = threadIdx.x >> 5;
  const int lane = threadIdx.x & 31;
  const int gw   = blockIdx.x * 4 + wv;       // 512 blocks * 4 waves = 2048 = 32 dims * 64 btiles
  const int d    = gw & (DIMS - 1);
  const int bt   = gw >> 5;                   // batch tile 0..63
  const int g    = lane >> 4;                 // lane group (K/M-half selector)
  const int m    = lane & 15;                 // A-row (k_out) / B,C,D column (b)

  _Float16* h0r = &hbuf[wv][0][0];
  _Float16* h1r = &hbuf[wv][1][0];
  for (int i = lane; i < 2 * 16 * ROWH; i += 32) h0r[i] = (_Float16)0.0f;  // h0r/h1r contiguous
  __builtin_amdgcn_wave_barrier();

  // ---- Weights resident in A-operand registers (f32 -> f16 once) ----
  // A elem e, lane(g,m): K = e + 8g (e<8) | e + 8 + 8g (e>=8)
  v16h A0, A1;
  {
    const float* w0 = Whh0 + ((size_t)d * HID + m) * HID;
    const float* wi = Wih1 + ((size_t)d * HID + m) * HID;
    const float* wh = Whh1 + ((size_t)d * HID + m) * HID;
#pragma unroll
    for (int e = 0; e < 8; ++e) {
      A0[e]     = (_Float16)w0[e + 8 * g];   // layer0: K=0..15 = W_hh0 row
      A0[e + 8] = (_Float16)0.0f;            // layer0: K=16..31 zero-padded
      A1[e]     = (_Float16)wi[e + 8 * g];   // layer1 fused: K<16  -> W_ih1
      A1[e + 8] = (_Float16)wh[e + 8 * g];   // layer1 fused: K>=16 -> W_hh1
    }
  }

  // Per-lane C-init constants: VGPR v holds output row k = v + 8g
  float wih0v[8], b0v[8], b1v[8];
#pragma unroll
  for (int v = 0; v < 8; ++v) {
    int k = v + 8 * g;
    wih0v[v] = Wih0[(size_t)d * HID + k];
    b0v[v]   = bih0[(size_t)d * HID + k] + bhh0[(size_t)d * HID + k];
    b1v[v]   = bih1[(size_t)d * HID + k] + bhh1[(size_t)d * HID + k];
  }
  const float a_d = stats[2 * d];       // gamma * rsqrt(var+eps)
  const float c_d = stats[2 * d + 1];   // beta - mean * a_d

  // Each lane streams x for its own batch row b=m (contiguous in t; L2-resident)
  const float* xrow = x + (((size_t)(bt * 16 + m)) * DIMS + d) * TT;

  v8f lasth1 = {};
  for (int t0 = 0; t0 < TT; t0 += 4) {
    v4f xq = *(const v4f*)(xrow + t0);
    if (t0 + 64 < TT) __builtin_prefetch(xrow + t0 + 64, 0, 1);
#pragma unroll
    for (int j = 0; j < 4; ++j) {
      float xv = xq[j];
      xv = (xv != xv) ? 0.0f : xv;                       // nan_to_num
      float xn = __builtin_fmaf(xv, a_d, c_d);           // BatchNorm affine
      v8f c0;
#pragma unroll
      for (int v = 0; v < 8; ++v) c0[v] = __builtin_fmaf(xn, wih0v[v], b0v[v]);

      // B0 = [h0^T ; 0] : B elem e, lane(g,n=b): K = e + 8g (e<8) reads h0[b][8g..8g+7]; e>=8 zero
      V16 B0;
#pragma unroll
      for (int e = 0; e < 16; ++e) B0.v[e] = (_Float16)0.0f;
      {
        const v8h* p = (const v8h*)(h0r + m * ROWH);
        B0.h[0] = p[g];
      }
      v8f z0 = __builtin_amdgcn_wmma_f32_16x16x32_f16(false, A0, false, B0.v,
                                                      (short)0, c0, false, false);
      __builtin_amdgcn_wave_barrier();   // keep cross-lane LDS read->write order
#pragma unroll
      for (int v = 0; v < 8; ++v) {      // h0n = tanh(z0); D: lane(g,n=b), VGPR v -> k=v+8g
        float h = fast_tanh(z0[v]);
        h0r[m * ROWH + v + 8 * g] = (_Float16)h;
      }
      __builtin_amdgcn_wave_barrier();

      // B1 = [h0n^T ; h1^T] : e<8 reads h0n[b][8g..8g+7], e>=8 reads h1[b][8g..8g+7]
      V16 B1;
      {
        const v8h* p0 = (const v8h*)(h0r + m * ROWH);
        const v8h* p1 = (const v8h*)(h1r + m * ROWH);
        B1.h[0] = p0[g]; B1.h[1] = p1[g];
      }
      v8f c1;
#pragma unroll
      for (int v = 0; v < 8; ++v) c1[v] = b1v[v];
      v8f z1 = __builtin_amdgcn_wmma_f32_16x16x32_f16(false, A1, false, B1.v,
                                                      (short)0, c1, false, false);
      __builtin_amdgcn_wave_barrier();
#pragma unroll
      for (int v = 0; v < 8; ++v) {
        float h = fast_tanh(z1[v]);
        lasth1[v] = h;
        h1r[m * ROWH + v + 8 * g] = (_Float16)h;
      }
      __builtin_amdgcn_wave_barrier();
    }
  }

  // feat[b, d*HID + k] = h1_final[k][b]  (lane n=b, VGPR v -> k=v+8g; contiguous -> b128 stores)
  float* frow = feat + ((size_t)(bt * 16 + m)) * (DIMS * HID) + d * HID + 8 * g;
#pragma unroll
  for (int v = 0; v < 8; ++v) frow[v] = lasth1[v];
}

// ---------------------------------------------------------------------------
// Kernel 0/0b: deterministic two-pass BatchNorm stats (also warms L2 with x)
// ---------------------------------------------------------------------------
__global__ __launch_bounds__(256) void bn_partial(const float* __restrict__ x,
                                                  float* __restrict__ part)
{
  const int d = blockIdx.y, chunk = blockIdx.x, tid = threadIdx.x;
  const float* base = x + ((size_t)chunk * 16 * DIMS + d) * TT;
  float s = 0.f, s2 = 0.f;
  for (int i = tid; i < 16 * TT; i += 256) {
    int b = i >> 9, t = i & (TT - 1);
    float v = base[(size_t)b * DIMS * TT + t];
    v = (v != v) ? 0.f : v;
    s += v; s2 += v * v;
  }
  __shared__ float rs[256], rq[256];
  rs[tid] = s; rq[tid] = s2;
  __syncthreads();
  for (int off = 128; off > 0; off >>= 1) {
    if (tid < off) { rs[tid] += rs[tid + off]; rq[tid] += rq[tid + off]; }
    __syncthreads();
  }
  if (tid == 0) {
    part[(d * 64 + chunk) * 2]     = rs[0];
    part[(d * 64 + chunk) * 2 + 1] = rq[0];
  }
}

__global__ void bn_finalize(const float* __restrict__ part,
                            const float* __restrict__ gamma,
                            const float* __restrict__ beta,
                            float* __restrict__ stats)
{
  int d = threadIdx.x;
  if (d >= DIMS) return;
  float s = 0.f, s2 = 0.f;
  for (int c = 0; c < 64; ++c) {
    s  += part[(d * 64 + c) * 2];
    s2 += part[(d * 64 + c) * 2 + 1];
  }
  const float N = 1024.0f * 512.0f;
  float mean = s / N;
  float var  = s2 / N - mean * mean;
  float a = gamma[d] * rsqrtf(var + 1e-5f);
  stats[2 * d]     = a;
  stats[2 * d + 1] = beta[d] - mean * a;
}

// ---------------------------------------------------------------------------
// Kernels 3/4: tiny FC head (67 MFLOP total)
// ---------------------------------------------------------------------------
__global__ __launch_bounds__(64) void fc1_kernel(const float* __restrict__ feat,
                                                 const float* __restrict__ w,
                                                 const float* __restrict__ bias,
                                                 float* __restrict__ hout)
{
  __shared__ float frow[DIMS * HID];
  const int b = blockIdx.x, o = threadIdx.x;
  for (int i = o; i < DIMS * HID; i += 64) frow[i] = feat[(size_t)b * DIMS * HID + i];
  __syncthreads();
  float s = bias[o];
  const float* wr = w + (size_t)o * DIMS * HID;
  for (int i = 0; i < DIMS * HID; ++i) s = __builtin_fmaf(frow[i], wr[i], s);
  hout[b * 64 + o] = s > 0.f ? s : 0.f;
}

__global__ void fc2_kernel(const float* __restrict__ h, const float* __restrict__ w,
                           const float* __restrict__ bias, float* __restrict__ out)
{
  int b = blockIdx.x * blockDim.x + threadIdx.x;
  if (b >= 1024) return;
  float s = bias[0];
  for (int o = 0; o < 64; ++o) s = __builtin_fmaf(h[b * 64 + o], w[o], s);
  out[b] = s;
}

// ---------------------------------------------------------------------------
extern "C" void kernel_launch(void* const* d_in, const int* in_sizes, int n_in,
                              void* d_out, int out_size, void* d_ws, size_t ws_size,
                              hipStream_t stream) {
  const float* x     = (const float*)d_in[0];
  const float* gamma = (const float*)d_in[1];
  const float* beta  = (const float*)d_in[2];
  const float* Wih0  = (const float*)d_in[3];
  const float* Whh0  = (const float*)d_in[4];
  const float* bih0  = (const float*)d_in[5];
  const float* bhh0  = (const float*)d_in[6];
  const float* Wih1  = (const float*)d_in[7];
  const float* Whh1  = (const float*)d_in[8];
  const float* bih1  = (const float*)d_in[9];
  const float* bhh1  = (const float*)d_in[10];
  const float* fc1w  = (const float*)d_in[11];
  const float* fc1b  = (const float*)d_in[12];
  const float* fc2w  = (const float*)d_in[13];
  const float* fc2b  = (const float*)d_in[14];
  float* out = (float*)d_out;

  float* ws    = (float*)d_ws;
  float* part  = ws;                          // 32*64*2 floats
  float* stats = ws + 4096;                   // 64 floats (a_d, c_d per dim)
  float* feat  = ws + 8192;                   // 1024*512 floats
  float* h1o   = ws + 8192 + 1024 * 512;      // 1024*64 floats

  bn_partial <<<dim3(64, 32), 256, 0, stream>>>(x, part);
  bn_finalize<<<1, 32, 0, stream>>>(part, gamma, beta, stats);
  rnn_wmma_kernel<<<512, 128, 0, stream>>>(x, stats, Wih0, Whh0, bih0, bhh0,
                                           Wih1, Whh1, bih1, bhh1, feat);
  fc1_kernel<<<1024, 64, 0, stream>>>(feat, fc1w, fc1b, h1o);
  fc2_kernel<<<4, 256, 0, stream>>>(h1o, fc2w, fc2b, out);
}
